// RowSelfAttention_35107062677526
// MI455X (gfx1250) — matrix operand order repaired and checked by reading, not compiled, not run on hardware
//
#include <hip/hip_runtime.h>
#include <hip/hip_bf16.h>
#include <math.h>

// Row self-attention, MI455X (gfx1250, wave32).
// All GEMMs via v_wmma_f32_16x16x32_bf16 (f32 accumulate); f32->bf16
// conversion folded into LDS staging. 256-thread blocks, double-buffered
// LDS addressed by XOR'd offsets (loop kept un-unrolled so accumulators
// stay pinned), b128 LDS traffic, 32x64 output per wave, and
// sched_group_barrier-enforced DS-read/WMMA interleave.

#define R_ROWS 128
#define C_COLS 512
#define EMB    768
#define NHEADS 12
#define HDIM   64
#define MTOK   (R_ROWS * C_COLS)   // 65536 tokens

typedef __attribute__((ext_vector_type(16))) __bf16 v16bf;
typedef __attribute__((ext_vector_type(8)))  float  v8f;

union Frag { v16bf v; unsigned u[8]; uint4 q[2]; };

__device__ __forceinline__ unsigned short f2bf(float f) {
  union { float f; unsigned u; } x; x.f = f;
  unsigned r = x.u + 0x7FFFu + ((x.u >> 16) & 1u);  // round-to-nearest-even
  return (unsigned short)(r >> 16);
}
__device__ __forceinline__ unsigned pack2(unsigned short lo, unsigned short hi) {
  return (unsigned)lo | ((unsigned)hi << 16);
}

#define LDSW 20  // 16 data dwords + 4 pad; multiple of 4 keeps 16B alignment

// ---- LDS staging (256 threads stage an RT x 32 bf16 tile) -----------------

template <int RT>
__device__ __forceinline__ void stage_rm_f32(unsigned* tile, const float* src,
                                             size_t rs, int tid) {
  constexpr int TPR = 256 / RT;   // threads per row
  constexpr int DW  = 16 / TPR;   // dwords per thread
  int row = tid / TPR, part = tid % TPR;
  const float4* p = (const float4*)(src + (size_t)row * rs + part * (DW * 2));
  uint4* d = (uint4*)(tile + row * LDSW + part * DW);
#pragma unroll
  for (int j = 0; j < DW / 4; ++j) {
    float4 f0 = p[2 * j], f1 = p[2 * j + 1];
    uint4 o;
    o.x = pack2(f2bf(f0.x), f2bf(f0.y));
    o.y = pack2(f2bf(f0.z), f2bf(f0.w));
    o.z = pack2(f2bf(f1.x), f2bf(f1.y));
    o.w = pack2(f2bf(f1.z), f2bf(f1.w));
    d[j] = o;
  }
}

template <int RT>
__device__ __forceinline__ void stage_rm_bf16(unsigned* tile, const unsigned short* src,
                                              size_t rs, int tid) {
  constexpr int TPR = 256 / RT;
  constexpr int DW  = 16 / TPR;
  int row = tid / TPR, part = tid % TPR;
  const uint4* p = (const uint4*)(src + (size_t)row * rs + part * (DW * 2));
  uint4* d = (uint4*)(tile + row * LDSW + part * DW);
#pragma unroll
  for (int j = 0; j < DW / 4; ++j) d[j] = p[j];
}

// B stored [k][n]; stage transposed as tile[n][k-pairs] (f32 src, scaled).
template <int RT>
__device__ __forceinline__ void stage_tr_f32(unsigned* tile, const float* src,
                                             size_t ldb, float scale, int tid) {
  constexpr int TPR = 256 / RT;
  constexpr int DW  = 16 / TPR;
  int n = tid / TPR, part = tid % TPR;
#pragma unroll
  for (int i = 0; i < DW; ++i) {
    size_t k = (size_t)(part * DW + i) * 2;
    tile[n * LDSW + part * DW + i] =
        pack2(f2bf(src[k * ldb + n] * scale), f2bf(src[(k + 1) * ldb + n] * scale));
  }
}
template <int RT>
__device__ __forceinline__ void stage_tr_bf16(unsigned* tile, const unsigned short* src,
                                              size_t ldb, int tid) {
  constexpr int TPR = 256 / RT;
  constexpr int DW  = 16 / TPR;
  int n = tid / TPR, part = tid % TPR;
#pragma unroll
  for (int i = 0; i < DW; ++i) {
    size_t k = (size_t)(part * DW + i) * 2;
    tile[n * LDSW + part * DW + i] = pack2(src[k * ldb + n], src[(k + 1) * ldb + n]);
  }
}

// ---- fragment loads per CDNA5 wave32 WMMA VGPR layouts (b128 LDS reads) ---

// A (16x32 bf16): lanes 0-15 row M=lane, dwords {0..3, 8..11};
// lanes 16-31 same rows, dwords {4..7, 12..15}.
__device__ __forceinline__ void load_a(const unsigned* tile, int mOff, int lane, Frag& f) {
  const uint4* p = (const uint4*)(tile + (mOff + (lane & 15)) * LDSW + ((lane >> 4) << 2));
  f.q[0] = p[0];   // K 0-7   (or 8-15)
  f.q[1] = p[2];   // K 16-23 (or 24-31)
}
// B (32x16 bf16): lane holds column N=lane%16; lanes 0-15 dwords {0..3, 8..11},
// lanes 16-31 dwords {4..7, 12..15} (same K order as A).
__device__ __forceinline__ void load_b(const unsigned* tile, int nOff, int lane, Frag& f) {
  const uint4* p = (const uint4*)(tile + (nOff + (lane & 15)) * LDSW + ((lane >> 4) << 2));
  f.q[0] = p[0];
  f.q[1] = p[2];
}

__device__ __forceinline__ v8f wmma_bf16(const Frag& a, const Frag& b, v8f c) {
  return __builtin_amdgcn_wmma_f32_16x16x32_bf16(false, a.v, false, b.v,
                                                 (short)0, c, false, false);
}

// One k-step for a wave tile of (AF*16) x (BF*16).
// sched_group_barrier pattern keeps only one B-fragment live at a time:
//   [AF*2 ds_read (A)] then BF x ( [2 ds_read (B_t)] -> [AF WMMA] )
template <int AF, int BF>
__device__ __forceinline__ void mma_tile(const unsigned* aT, const unsigned* bT,
                                         int mOff, int nOff, int lane,
                                         v8f (&acc)[AF][BF]) {
  Frag a[AF];
#pragma unroll
  for (int x = 0; x < AF; ++x) load_a(aT, mOff + x * 16, lane, a[x]);
#pragma unroll
  for (int t = 0; t < BF; ++t) {
    Frag b; load_b(bT, nOff + t * 16, lane, b);
#pragma unroll
    for (int x = 0; x < AF; ++x) acc[x][t] = wmma_bf16(a[x], b, acc[x][t]);
  }
#if __has_builtin(__builtin_amdgcn_sched_group_barrier)
  __builtin_amdgcn_sched_group_barrier(0x100, AF * 2, 0);   // DS reads: A frags
#pragma unroll
  for (int t = 0; t < BF; ++t) {
    __builtin_amdgcn_sched_group_barrier(0x100, 2, 0);      // DS reads: B frag t
    __builtin_amdgcn_sched_group_barrier(0x008, AF, 0);     // WMMAs using B frag t
  }
#endif
}

// ---- kernel 1/5: dense projection GEMM (QKV and output projection) --------
// C[M,768] = A[M,768] @ W[768,768] * scaleB + bias*scaleB. Block tile
// 128x128, 8 waves, each wave 32x64; XOR-offset double-buffered LDS.
template <bool ABF16, bool OUTF32>
__global__ void __launch_bounds__(256)
proj_gemm(const void* __restrict__ Ap, const float* __restrict__ W,
          const float* __restrict__ bias, void* __restrict__ Cp, float scaleB) {
  __shared__ __align__(16) unsigned aT[2 * 128 * LDSW];
  __shared__ __align__(16) unsigned bT[2 * 128 * LDSW];
  constexpr unsigned ABUF = 128 * LDSW, BBUF = 128 * LDSW;
  const int tid = threadIdx.x, lane = tid & 31, wave = tid >> 5;
  const int wr = wave >> 1, wc = wave & 1;
  const int mBase = blockIdx.x * 128, nBase = blockIdx.y * 128;
  constexpr int NSTEPS = EMB / 32;   // 24

  v8f acc[2][4];
  const v8f zero = {0.f, 0.f, 0.f, 0.f, 0.f, 0.f, 0.f, 0.f};
#pragma unroll
  for (int x = 0; x < 2; ++x)
#pragma unroll
    for (int t = 0; t < 4; ++t) acc[x][t] = zero;

  auto stage = [&](unsigned* aDst, unsigned* bDst, int ks) {
    int k0 = ks * 32;
    if constexpr (ABF16)
      stage_rm_bf16<128>(aDst, (const unsigned short*)Ap + (size_t)mBase * EMB + k0, EMB, tid);
    else
      stage_rm_f32<128>(aDst, (const float*)Ap + (size_t)mBase * EMB + k0, EMB, tid);
    stage_tr_f32<128>(bDst, W + (size_t)k0 * EMB + nBase, EMB, scaleB, tid);
  };

  stage(aT, bT, 0);
  unsigned off = 0;
#pragma clang loop unroll(disable)
  for (int ks = 0; ks < NSTEPS - 1; ++ks) {   // single symmetric body
    __syncthreads();
    stage(aT + (off ^ ABUF), bT + (off ^ BBUF), ks + 1);
    mma_tile<2, 4>(aT + off, bT + off, wr * 32, wc * 64, lane, acc);
    off ^= ABUF;
  }
  __syncthreads();
  mma_tile<2, 4>(aT + off, bT + off, wr * 32, wc * 64, lane, acc);  // peeled tail

  const int colBase = lane & 15, rowAdd = (lane >> 4) * 8;
#pragma unroll
  for (int x = 0; x < 2; ++x)
#pragma unroll
    for (int t = 0; t < 4; ++t) {
      int n = nBase + wc * 64 + t * 16 + colBase;
      float bv = bias[n] * scaleB;
#pragma unroll
      for (int i = 0; i < 8; ++i) {
        int m = mBase + wr * 32 + x * 16 + rowAdd + i;
        float val = acc[x][t][i] + bv;
        if constexpr (OUTF32)
          ((float*)Cp)[(size_t)m * EMB + n] = val;
        else
          ((unsigned short*)Cp)[(size_t)m * EMB + n] = f2bf(val);
      }
    }
}

// ---- kernel 2/5: logits[h,i,j] = sum_{r,d} q*k (NT, K = R*D = 8192) -------
__global__ void __launch_bounds__(256)
logits_gemm(const unsigned short* __restrict__ q, const unsigned short* __restrict__ kM,
            float* __restrict__ logits) {
  __shared__ __align__(16) unsigned aT[2 * 128 * LDSW];
  __shared__ __align__(16) unsigned bT[2 * 128 * LDSW];
  constexpr unsigned ABUF = 128 * LDSW, BBUF = 128 * LDSW;
  const int tid = threadIdx.x, lane = tid & 31, wave = tid >> 5;
  const int wr = wave >> 1, wc = wave & 1;
  const int iBase = blockIdx.x * 128, jBase = blockIdx.y * 128, h = blockIdx.z;
  constexpr int NSTEPS = 2 * R_ROWS;   // 256

  v8f acc[2][4];
  const v8f zero = {0.f, 0.f, 0.f, 0.f, 0.f, 0.f, 0.f, 0.f};
#pragma unroll
  for (int x = 0; x < 2; ++x)
#pragma unroll
    for (int t = 0; t < 4; ++t) acc[x][t] = zero;

  auto stage = [&](unsigned* aDst, unsigned* bDst, int ks) {
    int r = ks >> 1, kk = (ks & 1) * 32;
    size_t base = ((size_t)r * C_COLS) * EMB + h * HDIM + kk;
    stage_rm_bf16<128>(aDst, q  + base + (size_t)iBase * EMB, EMB, tid);
    stage_rm_bf16<128>(bDst, kM + base + (size_t)jBase * EMB, EMB, tid);  // NT
  };

  stage(aT, bT, 0);
  unsigned off = 0;
#pragma clang loop unroll(disable)
  for (int ks = 0; ks < NSTEPS - 1; ++ks) {
    __syncthreads();
    stage(aT + (off ^ ABUF), bT + (off ^ BBUF), ks + 1);
    mma_tile<2, 4>(aT + off, bT + off, wr * 32, wc * 64, lane, acc);
    off ^= ABUF;
  }
  __syncthreads();
  mma_tile<2, 4>(aT + off, bT + off, wr * 32, wc * 64, lane, acc);

  float* outp = logits + (size_t)h * C_COLS * C_COLS;
  const int colBase = lane & 15, rowAdd = (lane >> 4) * 8;
#pragma unroll
  for (int x = 0; x < 2; ++x)
#pragma unroll
    for (int t = 0; t < 4; ++t) {
      int j = jBase + wc * 64 + t * 16 + colBase;
#pragma unroll
      for (int i = 0; i < 8; ++i) {
        int ii = iBase + wr * 32 + x * 16 + rowAdd + i;
        outp[(size_t)ii * C_COLS + j] = acc[x][t][i];
      }
    }
}

// ---- kernel 3/5: softmax over j for each (h,i) row ------------------------
__global__ void __launch_bounds__(128)
softmax_rows(const float* __restrict__ logits, float* __restrict__ probsF,
             unsigned short* __restrict__ probsBf) {
  const int rowid = blockIdx.x;           // h*C + i
  const int tid = threadIdx.x;
  const float* src = logits + (size_t)rowid * C_COLS;
  __shared__ float red[128];

  float v[4], mx = -INFINITY;
#pragma unroll
  for (int j = 0; j < 4; ++j) { v[j] = src[tid + j * 128]; mx = fmaxf(mx, v[j]); }
  red[tid] = mx; __syncthreads();
  for (int s = 64; s > 0; s >>= 1) {
    if (tid < s) red[tid] = fmaxf(red[tid], red[tid + s]);
    __syncthreads();
  }
  mx = red[0]; __syncthreads();

  float sum = 0.f;
#pragma unroll
  for (int j = 0; j < 4; ++j) { v[j] = __expf(v[j] - mx); sum += v[j]; }
  red[tid] = sum; __syncthreads();
  for (int s = 64; s > 0; s >>= 1) {
    if (tid < s) red[tid] += red[tid + s];
    __syncthreads();
  }
  float inv = 1.f / red[0];

#pragma unroll
  for (int j = 0; j < 4; ++j) {
    float p = v[j] * inv;
    size_t idx = (size_t)rowid * C_COLS + tid + j * 128;
    probsF[idx]  = p;
    probsBf[idx] = f2bf(p);
  }
}

// ---- kernel 4/5: context[r,i,h,d] = sum_j probs[h,i,j] * v[r,j,h,d] -------
// One (h,r) batch per blockIdx.z: M=512(i), N=64(d), K=512(j). Tile 128x64.
__global__ void __launch_bounds__(256)
context_gemm(const unsigned short* __restrict__ probs, const unsigned short* __restrict__ vM,
             unsigned short* __restrict__ ctx) {
  __shared__ __align__(16) unsigned aT[2 * 128 * LDSW];
  __shared__ __align__(16) unsigned bT[2 * 64 * LDSW];
  constexpr unsigned ABUF = 128 * LDSW, BBUF = 64 * LDSW;
  const int tid = threadIdx.x, lane = tid & 31, wave = tid >> 5;
  const int h = blockIdx.z / R_ROWS, r = blockIdx.z % R_ROWS;
  const int iBase = blockIdx.x * 128;
  const unsigned short* A = probs + (size_t)h * C_COLS * C_COLS;      // lda = C
  const unsigned short* B = vM + (size_t)r * C_COLS * EMB + h * HDIM; // B[j][d], ldb = E
  constexpr int NSTEPS = C_COLS / 32;   // 16

  v8f acc[1][4];
  const v8f zero = {0.f, 0.f, 0.f, 0.f, 0.f, 0.f, 0.f, 0.f};
#pragma unroll
  for (int t = 0; t < 4; ++t) acc[0][t] = zero;

  auto stage = [&](unsigned* aDst, unsigned* bDst, int ks) {
    int j0 = ks * 32;
    stage_rm_bf16<128>(aDst, A + (size_t)iBase * C_COLS + j0, C_COLS, tid);
    stage_tr_bf16<64>(bDst, B + (size_t)j0 * EMB, EMB, tid);
  };

  stage(aT, bT, 0);
  unsigned offA = 0, offB = 0;
#pragma clang loop unroll(disable)
  for (int ks = 0; ks < NSTEPS - 1; ++ks) {
    __syncthreads();
    stage(aT + (offA ^ ABUF), bT + (offB ^ BBUF), ks + 1);
    mma_tile<1, 4>(aT + offA, bT + offB, wave * 16, 0, lane, acc);
    offA ^= ABUF; offB ^= BBUF;
  }
  __syncthreads();
  mma_tile<1, 4>(aT + offA, bT + offB, wave * 16, 0, lane, acc);

  const int colBase = lane & 15, rowAdd = (lane >> 4) * 8;
#pragma unroll
  for (int t = 0; t < 4; ++t) {
    int d = t * 16 + colBase;
#pragma unroll
    for (int i = 0; i < 8; ++i) {
      int ii = iBase + wave * 16 + rowAdd + i;
      ctx[((size_t)r * C_COLS + ii) * EMB + h * HDIM + d] = f2bf(acc[0][t][i]);
    }
  }
}

// ---- host side ------------------------------------------------------------
extern "C" void kernel_launch(void* const* d_in, const int* in_sizes, int n_in,
                              void* d_out, int out_size, void* d_ws, size_t ws_size,
                              hipStream_t stream) {
  (void)in_sizes; (void)n_in; (void)out_size; (void)ws_size;
  const float* x  = (const float*)d_in[0];
  const float* Wq = (const float*)d_in[1];
  const float* bq = (const float*)d_in[2];
  const float* Wk = (const float*)d_in[3];
  const float* bk = (const float*)d_in[4];
  const float* Wv = (const float*)d_in[5];
  const float* bv = (const float*)d_in[6];
  const float* Wo = (const float*)d_in[7];
  const float* bo = (const float*)d_in[8];

  float* out      = (float*)d_out;                       // [M, E]
  float* probsOut = out + (size_t)MTOK * EMB;            // [H, C, C]

  // workspace: q,k,v,ctx (bf16, M*E each) + logits f32 + probs bf16 (~402 MB)
  unsigned short* qbf = (unsigned short*)d_ws;
  unsigned short* kbf = qbf + (size_t)MTOK * EMB;
  unsigned short* vbf = kbf + (size_t)MTOK * EMB;
  unsigned short* ctx = vbf + (size_t)MTOK * EMB;
  float* logits = (float*)(ctx + (size_t)MTOK * EMB);
  unsigned short* pbf = (unsigned short*)(logits + (size_t)NHEADS * C_COLS * C_COLS);

  const float scaling = (1.0f / sqrtf((float)HDIM)) / sqrtf((float)R_ROWS);

  dim3 blk256(256);
  dim3 gProj(MTOK / 128, EMB / 128);
  proj_gemm<false, false><<<gProj, blk256, 0, stream>>>(x, Wq, bq, qbf, scaling);
  proj_gemm<false, false><<<gProj, blk256, 0, stream>>>(x, Wk, bk, kbf, 1.0f);
  proj_gemm<false, false><<<gProj, blk256, 0, stream>>>(x, Wv, bv, vbf, 1.0f);

  dim3 gLog(C_COLS / 128, C_COLS / 128, NHEADS);
  logits_gemm<<<gLog, blk256, 0, stream>>>(qbf, kbf, logits);

  softmax_rows<<<dim3(NHEADS * C_COLS), dim3(128), 0, stream>>>(logits, probsOut, pbf);

  dim3 gCtx(C_COLS / 128, 1, NHEADS * R_ROWS);
  context_gemm<<<gCtx, blk256, 0, stream>>>(pbf, vbf, ctx);

  proj_gemm<true, true><<<gProj, blk256, 0, stream>>>(ctx, Wo, bo, out, 1.0f);
}
